// MultiHeadAttentionV2_37546604102439
// MI455X (gfx1250) — matrix operand order repaired and checked by reading, not compiled, not run on hardware
//
#include <hip/hip_runtime.h>
#include <hip/hip_bf16.h>
#include <stdint.h>

// ---------------------------------------------------------------------------
// MI455X (gfx1250) fused MHA-variant:
//   Q = x @ Wq (bf16 WMMA GEMM, async-to-LDS staging)
//   flash attention per (b,h): S = Q xT / sqrt(512), online softmax, O = P V
//   out = attn @ Wp + bp (bf16 WMMA GEMM, f32 epilogue)
// ---------------------------------------------------------------------------

typedef __bf16 bf16;
typedef __attribute__((ext_vector_type(16))) __bf16 v16bf;
typedef __attribute__((ext_vector_type(8)))  float  v8f;
typedef __attribute__((ext_vector_type(8)))  uint32_t v8u;

#define NB    2048   // sequence length
#define DH    512    // per-head dim (== D)
#define NH    8      // heads
#define HDIM  4096   // H * D
#define MROWS 8192   // B * N

static __device__ inline v8f wmma_bf16(v16bf a, v16bf b, v8f c) {
  return __builtin_amdgcn_wmma_f32_16x16x32_bf16(false, a, false, b, (short)0, c,
                                                 false, false);
}

// CDNA5 async global->LDS copy, 16B per lane, tracked by ASYNCcnt.
// (inline asm: portable across ROCm7.2 / amdgpu-toolchain builtin arity skew)
static __device__ inline void async_copy_b128(uint32_t lds_off, const void* gptr) {
  uint64_t ga = (uint64_t)(uintptr_t)gptr;
  asm volatile("global_load_async_to_lds_b128 %0, %1, off"
               :: "v"(lds_off), "v"(ga) : "memory");
}
static __device__ inline void wait_asynccnt0() {
  asm volatile("s_wait_asynccnt 0" ::: "memory");
}
// generic LDS pointer -> raw LDS byte address (flat LDS addr = addr[31:0])
static __device__ inline uint32_t lds_addr(const void* p) {
  return (uint32_t)(uintptr_t)p;
}

// A fragment 16(M)x32(K) bf16, tile row-major [rows][ldk], pairs K-contiguous.
static __device__ inline v16bf load_a_frag(const bf16* tile, int ldk) {
  int lane = threadIdx.x & 31;
  int m  = lane & 15;
  int hi = (lane >> 4) * 16;
  const bf16* p = tile + m * ldk;
  v8u u;
#pragma unroll
  for (int v = 0; v < 8; ++v) {
    int k0 = 2 * v + hi;
    u[v] = *(const uint32_t*)(p + k0);
  }
  return __builtin_bit_cast(v16bf, u);
}

// B fragment 32(K)x16(N) bf16; tile stored [n][ldk] so K pairs are contiguous.
static __device__ inline v16bf load_b_frag(const bf16* tile, int ldk) {
  int lane = threadIdx.x & 31;
  int n  = lane & 15;
  int kb = (lane >> 4) * 16;
  const bf16* p = tile + n * ldk + kb;
  v8u u;
#pragma unroll
  for (int v = 0; v < 8; ++v) u[v] = *(const uint32_t*)(p + 2 * v);
  return __builtin_bit_cast(v16bf, u);
}

// ---------------------------------------------------------------------------
__global__ void cvt_bf16(const float* __restrict__ in, bf16* __restrict__ out, int n) {
  int i = blockIdx.x * 256 + threadIdx.x;
  if (i < n) out[i] = (bf16)in[i];
}

// ---------------------------------------------------------------------------
// C[M,N] = A[M,K] @ Bw[K,N]; tile 128x128, BK=32, 256 threads = 8 waves (2x4),
// each wave computes 64x32 via 4x2 WMMA fragments.
// mode 0: bf16 output to obf. mode 1: f32 output + bias to of32.
__global__ __launch_bounds__(256) void gemm_wmma(const bf16* __restrict__ A,
                                                 const bf16* __restrict__ Bw,
                                                 int N, int K,
                                                 bf16* __restrict__ obf,
                                                 float* __restrict__ of32,
                                                 const float* __restrict__ bias,
                                                 int mode) {
  extern __shared__ char smem[];
  bf16* As = (bf16*)smem;                  // [128][32]
  bf16* Bs = (bf16*)(smem + 128 * 32 * 2); // [128 n][32 k]

  int tid  = threadIdx.x;
  int lane = tid & 31;
  int wave = tid >> 5;
  int wm = wave >> 2;          // 0..1  (64 rows each)
  int wn = wave & 3;           // 0..3  (32 cols each)
  int tileM = blockIdx.y * 128;
  int tileN = blockIdx.x * 128;

  v8f c[4][2] = {};

  for (int k0 = 0; k0 < K; k0 += 32) {
    // stage A tile via async global->LDS (2 x b128 per thread, 16 bf16)
    {
      int r = tid >> 1, ch = (tid & 1) * 16;
      const bf16* src = A + (size_t)(tileM + r) * K + k0 + ch;
      uint32_t dst = lds_addr(As + r * 32 + ch);
      async_copy_b128(dst,     src);
      async_copy_b128(dst + 16, src + 8);
    }
    // stage B tile transposed into [n][k] (needs element transpose -> manual)
    {
      int kk = tid >> 3, n0 = (tid & 7) * 16;
      const bf16* src = Bw + (size_t)(k0 + kk) * N + tileN + n0;
#pragma unroll
      for (int i = 0; i < 16; ++i) Bs[(n0 + i) * 32 + kk] = src[i];
    }
    if (k0 + 32 < K) {
      __builtin_prefetch(A + (size_t)(tileM + (tid >> 1)) * K + k0 + 32, 0, 1);
      __builtin_prefetch(Bw + (size_t)(k0 + 32 + (tid >> 3)) * N + tileN, 0, 1);
    }
    wait_asynccnt0();
    __syncthreads();

    v16bf a[4], b[2];
#pragma unroll
    for (int i = 0; i < 4; ++i) a[i] = load_a_frag(As + (wm * 64 + i * 16) * 32, 32);
#pragma unroll
    for (int j = 0; j < 2; ++j) b[j] = load_b_frag(Bs + (wn * 32 + j * 16) * 32, 32);
#pragma unroll
    for (int i = 0; i < 4; ++i)
#pragma unroll
      for (int j = 0; j < 2; ++j) c[i][j] = wmma_bf16(a[i], b[j], c[i][j]);
    __syncthreads();
  }

  // epilogue: C/D layout: n = lane&15, m = v + 8*(lane>>4)
  int nloc = lane & 15;
  int mhi  = (lane >> 4) * 8;
#pragma unroll
  for (int i = 0; i < 4; ++i)
#pragma unroll
    for (int j = 0; j < 2; ++j) {
      int col = tileN + wn * 32 + j * 16 + nloc;
#pragma unroll
      for (int v = 0; v < 8; ++v) {
        int row = tileM + wm * 64 + i * 16 + mhi + v;
        float val = c[i][j][v];
        if (mode == 0) obf[(size_t)row * N + col] = (bf16)val;
        else           of32[(size_t)row * N + col] = val + bias[col];
      }
    }
}

// ---------------------------------------------------------------------------
// Fused flash attention. One WG = (b, h, 32-query tile), 256 threads = 8 waves.
// Waves split the 512-wide head dim for O (64 dims each, O in registers) and
// split the K=512 score contraction (partial S summed in LDS via ds_add_f32).
__global__ __launch_bounds__(256) void attn_kernel(const bf16* __restrict__ xbf,
                                                   const bf16* __restrict__ qbf,
                                                   bf16* __restrict__ attn) {
  extern __shared__ char smem[];
  bf16*  Qs = (bf16*)smem;                 // [32][512]
  bf16*  Ks = Qs + 32 * DH;                // [32][512]
  bf16*  Vt = Ks + 32 * DH;                // [512][32] transposed
  float* Sl = (float*)(Vt + DH * 32);      // [32][32]
  bf16*  Pl = (bf16*)(Sl + 32 * 32);       // [32][32]
  float* rmax = (float*)(Pl + 32 * 32);    // [32]
  float* rsum = rmax + 32;                 // [32]
  float* ralf = rsum + 32;                 // [32]

  int tid  = threadIdx.x;
  int lane = tid & 31;
  int wave = tid >> 5;
  int nloc = lane & 15;
  int mhi  = (lane >> 4) * 8;

  int idx = blockIdx.x;
  int qt = idx & 63;            // 64 query tiles of 32
  int bh = idx >> 6;
  int h  = bh & 7;
  int b  = bh >> 3;
  int q0 = qt * 32;
  int d0 = wave * 64;           // this wave's output-dim slice
  const float scale = 0.044194173824159216f;  // 1/sqrt(512)

  // stage Q tile (rows q0..q0+31, head slice) via async global->LDS
  {
    int r = tid >> 3, c0 = (tid & 7) * 64;
    const bf16* src = qbf + (size_t)(b * NB + q0 + r) * HDIM + h * DH + c0;
    uint32_t dst = lds_addr(Qs + r * DH + c0);
#pragma unroll
    for (int i = 0; i < 8; ++i) async_copy_b128(dst + i * 16, src + i * 8);
  }
  if (tid < 32) { rmax[tid] = -1e30f; rsum[tid] = 0.0f; }
  v8f o[2][4] = {};
  wait_asynccnt0();
  __syncthreads();

  for (int kb = 0; kb < NB; kb += 32) {
    // zero partial-S accumulator
    for (int i = tid; i < 32 * 32; i += 256) Sl[i] = 0.0f;
    // stage K tile (x rows kb..kb+31) via async global->LDS
    {
      int r = tid >> 3, c0 = (tid & 7) * 64;
      const bf16* src = xbf + (size_t)(b * NB + kb + r) * DH + c0;
      uint32_t dst = lds_addr(Ks + r * DH + c0);
#pragma unroll
      for (int i = 0; i < 8; ++i) async_copy_b128(dst + i * 16, src + i * 8);
    }
    // stage V tile transposed (V = Q-projection rows kb..kb+31)
    {
      int r = tid >> 3, c0 = (tid & 7) * 64;
      const bf16* src = qbf + (size_t)(b * NB + kb + r) * HDIM + h * DH + c0;
#pragma unroll
      for (int i = 0; i < 64; ++i) Vt[(c0 + i) * 32 + r] = src[i];
    }
    wait_asynccnt0();
    __syncthreads();

    // partial S over this wave's K slice [wave*64, wave*64+64)
    int ks0 = wave * 64;
    v8f s[2][2] = {};
#pragma unroll
    for (int kk = 0; kk < 2; ++kk)
#pragma unroll
      for (int qf = 0; qf < 2; ++qf) {
        v16bf aq = load_a_frag(Qs + (qf * 16) * DH + ks0 + kk * 32, DH);
#pragma unroll
        for (int kf = 0; kf < 2; ++kf) {
          v16bf bk = load_b_frag(Ks + (kf * 16) * DH + ks0 + kk * 32, DH);
          s[qf][kf] = wmma_bf16(aq, bk, s[qf][kf]);
        }
      }
    // reduce partial S across waves (ds_add_f32)
#pragma unroll
    for (int qf = 0; qf < 2; ++qf)
#pragma unroll
      for (int kf = 0; kf < 2; ++kf)
#pragma unroll
        for (int v = 0; v < 8; ++v)
          atomicAdd(&Sl[(qf * 16 + mhi + v) * 32 + kf * 16 + nloc],
                    s[qf][kf][v] * scale);
    __syncthreads();

    // online softmax (one thread per query row)
    if (tid < 32) {
      int r = tid;
      float bm = -1e30f;
      for (int j = 0; j < 32; ++j) bm = fmaxf(bm, Sl[r * 32 + j]);
      float om = rmax[r];
      float nm = fmaxf(om, bm);
      float al = __expf(om - nm);
      float sum = rsum[r] * al;
      for (int j = 0; j < 32; ++j) {
        float p = __expf(Sl[r * 32 + j] - nm);
        sum += p;
        Pl[r * 32 + j] = (bf16)p;
      }
      rmax[r] = nm; rsum[r] = sum; ralf[r] = al;
    }
    __syncthreads();

    // rescale O and accumulate P @ V for this wave's dim slice
#pragma unroll
    for (int qf = 0; qf < 2; ++qf) {
      float av[8];
#pragma unroll
      for (int v = 0; v < 8; ++v) av[v] = ralf[qf * 16 + mhi + v];
#pragma unroll
      for (int df = 0; df < 4; ++df)
#pragma unroll
        for (int v = 0; v < 8; ++v) o[qf][df][v] *= av[v];
      v16bf pa = load_a_frag(Pl + (qf * 16) * 32, 32);
#pragma unroll
      for (int df = 0; df < 4; ++df) {
        v16bf vb = load_b_frag(Vt + (d0 + df * 16) * 32, 32);
        o[qf][df] = wmma_bf16(pa, vb, o[qf][df]);
      }
    }
    __syncthreads();
  }

  // normalize and store into [B*N, HDIM] concat layout
#pragma unroll
  for (int qf = 0; qf < 2; ++qf) {
    float inv[8];
#pragma unroll
    for (int v = 0; v < 8; ++v) inv[v] = 1.0f / rsum[qf * 16 + mhi + v];
#pragma unroll
    for (int df = 0; df < 4; ++df)
#pragma unroll
      for (int v = 0; v < 8; ++v) {
        int qrow = q0 + qf * 16 + mhi + v;
        int dim  = d0 + df * 16 + nloc;
        attn[(size_t)(b * NB + qrow) * HDIM + h * DH + dim] =
            (bf16)(o[qf][df][v] * inv[v]);
      }
  }
}

// ---------------------------------------------------------------------------
extern "C" void kernel_launch(void* const* d_in, const int* in_sizes, int n_in,
                              void* d_out, int out_size, void* d_ws, size_t ws_size,
                              hipStream_t stream) {
  (void)in_sizes; (void)n_in; (void)out_size; (void)ws_size;
  const float* x  = (const float*)d_in[0];   // [4,2048,512]
  const float* Wq = (const float*)d_in[1];   // [512,4096]
  const float* Wp = (const float*)d_in[3];   // [4096,512]
  const float* bp = (const float*)d_in[4];   // [512]
  float* out = (float*)d_out;                // [4,2048,512]

  bf16* xbf  = (bf16*)d_ws;                                  // 8192*512
  bf16* wqbf = xbf  + (size_t)MROWS * DH;                    // 512*4096
  bf16* wpbf = wqbf + (size_t)DH * HDIM;                     // 4096*512
  bf16* qbf  = wpbf + (size_t)HDIM * DH;                     // 8192*4096
  bf16* abf  = qbf  + (size_t)MROWS * HDIM;                  // 8192*4096

  const int nX = MROWS * DH, nWq = DH * HDIM, nWp = HDIM * DH;
  cvt_bf16<<<(nX  + 255) / 256, 256, 0, stream>>>(x,  xbf,  nX);
  cvt_bf16<<<(nWq + 255) / 256, 256, 0, stream>>>(Wq, wqbf, nWq);
  cvt_bf16<<<(nWp + 255) / 256, 256, 0, stream>>>(Wp, wpbf, nWp);

  const int GEMM_LDS = 2 * 128 * 32 * 2;  // A + B tiles, bf16
  // Q = x @ Wq  (8192 x 4096, K=512), bf16 out
  gemm_wmma<<<dim3(HDIM / 128, MROWS / 128), 256, GEMM_LDS, stream>>>(
      xbf, wqbf, HDIM, DH, qbf, nullptr, nullptr, 0);

  // fused attention
  const int ATTN_LDS = 32 * DH * 2       // Qs
                     + 32 * DH * 2       // Ks
                     + DH * 32 * 2       // Vt
                     + 32 * 32 * 4       // Sl
                     + 32 * 32 * 2       // Pl
                     + 3 * 32 * 4;       // rmax/rsum/alpha
  attn_kernel<<<4 * NH * (NB / 32), 256, ATTN_LDS, stream>>>(xbf, qbf, abf);

  // out = attn @ Wp + bp  (8192 x 512, K=4096), f32 out
  gemm_wmma<<<dim3(DH / 128, MROWS / 128), 256, GEMM_LDS, stream>>>(
      abf, wpbf, DH, HDIM, nullptr, out, bp, 1);
}
